// AttentionDecoder_46591805227651
// MI455X (gfx1250) — matrix operand order repaired and checked by reading, not compiled, not run on hardware
//
#include <hip/hip_runtime.h>
#include <hip/hip_bf16.h>
#include <stdint.h>

// ---------------------------------------------------------------------------
// Problem constants (from reference)
// ---------------------------------------------------------------------------
#define BB      128
#define T_ENC   64
#define ENC_H   1024
#define EMB     512
#define HID     512
#define ALIGN_A 256
#define VOCAB   20000
#define NSTEP   16

typedef __attribute__((ext_vector_type(16))) __bf16 v16bf;
typedef __attribute__((ext_vector_type(8)))  float  v8f;
typedef __attribute__((ext_vector_type(4)))  int    v4i_;

struct Frag128x2 { uint4 lo, hi; };   // 32 bytes == v16bf

// round-to-nearest-even f32 -> bf16 (stored as ushort)
__device__ __forceinline__ unsigned short f2bf(float f) {
    unsigned int u = __builtin_bit_cast(unsigned int, f);
    u += 0x7FFFu + ((u >> 16) & 1u);
    return (unsigned short)(u >> 16);
}

__device__ __forceinline__ float sigm(float x) { return 1.0f / (1.0f + __expf(-x)); }

// ---------------------------------------------------------------------------
// CDNA5 async global->LDS copy (ASYNCcnt) with safe fallback.
// Probe-confirmed prototype: (v4i AS1* src, v4i AS3* dst, imm offset, imm cpol)
// ---------------------------------------------------------------------------
__device__ __forceinline__ void async_copy_b128(const void* g, void* l) {
#if __has_builtin(__builtin_amdgcn_global_load_async_to_lds_b128)
    __builtin_amdgcn_global_load_async_to_lds_b128(
        (__attribute__((address_space(1))) v4i_*)g,
        (__attribute__((address_space(3))) v4i_*)l,
        0, 0);
#else
    *(uint4*)l = *(const uint4*)g;      // global_load_b128 + ds_store_b128
#endif
}

__device__ __forceinline__ void wait_async0() {
#if __has_builtin(__builtin_amdgcn_global_load_async_to_lds_b128)
#if __has_builtin(__builtin_amdgcn_s_wait_asynccnt)
    __builtin_amdgcn_s_wait_asynccnt(0);
#else
    asm volatile("s_wait_asynccnt 0" ::: "memory");
#endif
#endif
}

// ---------------------------------------------------------------------------
// A-fragment: 16x32 bf16, lanes 0-15 row M=lane K={k0..k0+7, k0+16..k0+23},
// lanes 16-31 same rows, K shifted by +8  (ISA 7.12.2, 16-bit A 16x32)
// ---------------------------------------------------------------------------
__device__ __forceinline__ v16bf load_frag_a(const unsigned short* A, long lda,
                                             int m0, int k0, int lane) {
    int lrow = lane & 15, hi = lane >> 4;
    const unsigned short* p = A + (long)(m0 + lrow) * lda + k0 + hi * 8;
    Frag128x2 f;
    f.lo = *(const uint4*)(p);        // K = k0+hi*8    .. +7
    f.hi = *(const uint4*)(p + 16);   // K = k0+16+hi*8 .. +7
    return __builtin_bit_cast(v16bf, f);
}

// ---------------------------------------------------------------------------
// Block-cooperative bf16 WMMA GEMM:
//   C(MxN,f32) = act( A(MxK) * W(N,K)^T + bias[N] )
// block = 256 thr = 8 waves; macro tile = 128(M) x 64(N); K step 32.
// B panel (64 rows x 32 k = 4KB) double-buffered in LDS via async-to-LDS;
// each wave then reads its fragments with ds_load_b128 (conflict-free pad).
// Requires: M % 128 == 0, K % 32 == 0. N handled per-16 tile on stores.
// ---------------------------------------------------------------------------
#define BPAD 40   // row pitch in shorts: 80B -> 16B aligned, bank-conflict-free

__global__ __launch_bounds__(256)
void wmma_gemm_bf16(const unsigned short* __restrict__ A, long lda,
                    const unsigned short* __restrict__ W, long ldw,
                    float* __restrict__ C, long ldc,
                    const float* __restrict__ bias,
                    int M, int N, int K, int act /*0=none,1=relu*/) {
    __shared__ unsigned short sB[2][64][BPAD];

    const int tid  = threadIdx.x;
    const int lane = tid & 31;
    const int wave = __builtin_amdgcn_readfirstlane(tid >> 5);   // scalar
    const int grpsN = (N + 63) >> 6;
    const int m0 = (blockIdx.x / grpsN) * 128 + wave * 16;
    const int n0 = __builtin_amdgcn_readfirstlane((blockIdx.x % grpsN) * 64);

    // cooperative B panel load: thread -> (row 0..63, k-chunk of 8)
    const int brow = tid >> 2;
    const int bchk = (tid & 3) * 8;
    const int wr   = min(n0 + brow, N - 1);                // clamp OOB rows
    const unsigned short* wsrc = W + (long)wr * ldw + bchk;

    // prologue: panel k0 = 0 into buffer 0
    async_copy_b128(wsrc, &sB[0][brow][bchk]);

    v8f acc0 = {}, acc1 = {}, acc2 = {}, acc3 = {};
    const int lcol = lane & 15, hi = lane >> 4;
    int cur = 0;

    for (int k0 = 0; k0 < K; k0 += 32) {
        wait_async0();          // my panel chunks have landed in LDS
        __syncthreads();        // everyone's chunks landed; prev compute done

        if (k0 + 32 < K)        // prefetch next panel into the other buffer
            async_copy_b128(wsrc + k0 + 32, &sB[cur ^ 1][brow][bchk]);

        v16bf a = load_frag_a(A, lda, m0, k0, lane);
        if (k0 + 32 < K)
            __builtin_prefetch(A + (long)(m0 + lcol) * lda + k0 + 32, 0, 1);

        // B fragment nt: rows n0+nt*16+lcol, K half selected by lane-half
        #pragma unroll
        for (int nt = 0; nt < 4; ++nt) {
            const uint4* p =
                (const uint4*)&sB[cur][nt * 16 + lcol][hi * 8];
            Frag128x2 f; f.lo = p[0]; f.hi = p[2];
            v16bf b = __builtin_bit_cast(v16bf, f);
            switch (nt) {
            case 0: acc0 = __builtin_amdgcn_wmma_f32_16x16x32_bf16(
                        false, a, false, b, (short)0, acc0, false, false); break;
            case 1: acc1 = __builtin_amdgcn_wmma_f32_16x16x32_bf16(
                        false, a, false, b, (short)0, acc1, false, false); break;
            case 2: acc2 = __builtin_amdgcn_wmma_f32_16x16x32_bf16(
                        false, a, false, b, (short)0, acc2, false, false); break;
            case 3: acc3 = __builtin_amdgcn_wmma_f32_16x16x32_bf16(
                        false, a, false, b, (short)0, acc3, false, false); break;
            }
        }
        cur ^= 1;
    }

    // store: VGPR r -> row m0 + 8*hi + r, col n0 + nt*16 + lcol
    v8f accs[4] = {acc0, acc1, acc2, acc3};
    #pragma unroll
    for (int nt = 0; nt < 4; ++nt) {
        int n = n0 + nt * 16 + lcol;
        if (n < N) {                       // store-only predication (no WMMA)
            float bv = bias ? bias[n] : 0.0f;
            #pragma unroll
            for (int r = 0; r < 8; ++r) {
                float v = accs[nt][r] + bv;
                if (act == 1) v = fmaxf(v, 0.0f);
                C[(long)(m0 + hi * 8 + r) * ldc + n] = v;
            }
        }
    }
}

// ---------------------------------------------------------------------------
// f32 -> bf16 strided 2-D converter
// ---------------------------------------------------------------------------
__global__ void cvt2d_bf16(const float* __restrict__ src, long sstride,
                           unsigned short* __restrict__ dst, long dstride,
                           long rows, long cols) {
    long i = (long)blockIdx.x * blockDim.x + threadIdx.x;
    if (i >= rows * cols) return;
    long r = i / cols, c = i % cols;
    dst[r * dstride + c] = f2bf(src[r * sstride + c]);
}

__global__ void add_vec(const float* a, const float* b, float* o, int n) {
    int i = blockIdx.x * blockDim.x + threadIdx.x;
    if (i < n) o[i] = a[i] + b[i];
}

// ---------------------------------------------------------------------------
// init: split mapped(128x1024) -> h,c ; build hc_bf16 and h-slot of a_cat
// ---------------------------------------------------------------------------
__global__ void init_state(const float* __restrict__ mapped,
                           float* __restrict__ h, float* __restrict__ c,
                           unsigned short* __restrict__ hc_bf,
                           unsigned short* __restrict__ a_cat) {
    int i = blockIdx.x * blockDim.x + threadIdx.x;   // 0 .. 128*512-1
    if (i >= BB * HID) return;
    int b = i >> 9, j = i & 511;
    float hv = mapped[b * 1024 + j];
    float cv = mapped[b * 1024 + 512 + j];
    h[i] = hv;  c[i] = cv;
    hc_bf[b * 1024 + j]       = f2bf(hv);
    hc_bf[b * 1024 + 512 + j] = f2bf(cv);
    a_cat[(long)b * 2048 + 1536 + j] = f2bf(hv);
}

// ---------------------------------------------------------------------------
// Attention + context + embedding gather for one step.
// grid = 128 blocks (one per batch row), block = 256 threads.
// ---------------------------------------------------------------------------
__global__ __launch_bounds__(256)
void attention_step(const float* __restrict__ enc,       // (B,T,ENC_H)
                    const float* __restrict__ enc_proj,  // (B,T,ALIGN)
                    const float* __restrict__ salign,    // (B,ALIGN)
                    const float* __restrict__ wp,        // (ALIGN)
                    const float* __restrict__ emb,       // (VOCAB,EMB)
                    const int*   __restrict__ captions,  // (B,NSTEP)
                    int tstep,
                    unsigned short* __restrict__ a_cat)  // (B,2048) bf16
{
    const int b = blockIdx.x, tid = threadIdx.x;
    __shared__ float sal[ALIGN_A];
    __shared__ float part[T_ENC][4];
    __shared__ float sc[T_ENC];
    __shared__ float red[2];

    sal[tid] = salign[b * ALIGN_A + tid];
    __syncthreads();

    // scores[t] = sum_a tanh(enc_proj[b,t,a] + sal[a]) * wp[a]
    {
        int t = tid >> 2, sub = tid & 3;
        const float* ep = enc_proj + ((long)b * T_ENC + t) * ALIGN_A + sub * 64;
        float s = 0.0f;
        #pragma unroll 8
        for (int a = 0; a < 64; ++a)
            s += tanhf(ep[a] + sal[sub * 64 + a]) * wp[sub * 64 + a];
        part[t][sub] = s;
    }
    __syncthreads();
    if (tid < T_ENC)
        sc[tid] = part[tid][0] + part[tid][1] + part[tid][2] + part[tid][3];
    __syncthreads();
    if (tid == 0) {
        float mx = -3.4e38f;
        for (int i = 0; i < T_ENC; ++i) mx = fmaxf(mx, sc[i]);
        float sum = 0.0f;
        for (int i = 0; i < T_ENC; ++i) sum += __expf(sc[i] - mx);
        red[0] = mx; red[1] = sum;
    }
    __syncthreads();
    if (tid < T_ENC) sc[tid] = __expf(sc[tid] - red[0]) / red[1];
    __syncthreads();

    // ctx[e] = sum_t attn[t] * enc[b,t,e]  -> bf16 into a_cat[:,0:1024]
    #pragma unroll
    for (int q = 0; q < 4; ++q) {
        int e = tid + q * 256;
        const float* eb = enc + (long)b * T_ENC * ENC_H + e;
        float acc = 0.0f;
        #pragma unroll 8
        for (int t2 = 0; t2 < T_ENC; ++t2) acc += sc[t2] * eb[(long)t2 * ENC_H];
        a_cat[(long)b * 2048 + e] = f2bf(acc);
    }

    // embedding gather -> bf16 into a_cat[:,1024:1536]
    int tok = captions[b * NSTEP + tstep];
    #pragma unroll
    for (int q = 0; q < 2; ++q) {
        int j = tid + q * 256;
        a_cat[(long)b * 2048 + 1024 + j] = f2bf(emb[(long)tok * EMB + j]);
    }
}

// ---------------------------------------------------------------------------
// LSTM cell elementwise; refreshes h,c (f32), hc_bf16 and h-slot of a_cat
// ---------------------------------------------------------------------------
__global__ void lstm_cell(const float* __restrict__ gates,   // (B,2048) i|f|g|o
                          float* __restrict__ h, float* __restrict__ c,
                          unsigned short* __restrict__ hc_bf,
                          unsigned short* __restrict__ a_cat) {
    int i = blockIdx.x * blockDim.x + threadIdx.x;
    if (i >= BB * HID) return;
    int b = i >> 9, j = i & 511;
    const float* g = gates + (long)b * 2048;
    float ig = sigm(g[j]);
    float fg = sigm(g[512 + j]);
    float gg = tanhf(g[1024 + j]);
    float og = sigm(g[1536 + j]);
    float cn = fg * c[i] + ig * gg;
    float hn = og * tanhf(cn);
    c[i] = cn; h[i] = hn;
    hc_bf[b * 1024 + j]       = f2bf(hn);
    hc_bf[b * 1024 + 512 + j] = f2bf(cn);
    a_cat[(long)b * 2048 + 1536 + j] = f2bf(hn);
}

// ---------------------------------------------------------------------------
// log_softmax over VOCAB, write into d_out[b, tstep, :]
// ---------------------------------------------------------------------------
__global__ __launch_bounds__(256)
void log_softmax_out(const float* __restrict__ logits, float* __restrict__ out,
                     int tstep) {
    const int b = blockIdx.x, tid = threadIdx.x;
    const float* row = logits + (long)b * VOCAB;
    __shared__ float red[256];

    float mx = -3.4e38f;
    for (int i = tid; i < VOCAB; i += 256) mx = fmaxf(mx, row[i]);
    red[tid] = mx; __syncthreads();
    for (int s = 128; s > 0; s >>= 1) {
        if (tid < s) red[tid] = fmaxf(red[tid], red[tid + s]);
        __syncthreads();
    }
    mx = red[0]; __syncthreads();

    float sum = 0.0f;
    for (int i = tid; i < VOCAB; i += 256) sum += __expf(row[i] - mx);
    red[tid] = sum; __syncthreads();
    for (int s = 128; s > 0; s >>= 1) {
        if (tid < s) red[tid] += red[tid + s];
        __syncthreads();
    }
    float lse = mx + logf(red[0]);

    float* o = out + ((long)b * NSTEP + tstep) * VOCAB;
    for (int i = tid; i < VOCAB; i += 256) o[i] = row[i] - lse;
}

// ---------------------------------------------------------------------------
// Host side
// ---------------------------------------------------------------------------
static inline void launch_gemm(const unsigned short* A, long lda,
                               const unsigned short* W, long ldw,
                               float* C, long ldc, const float* bias,
                               int M, int N, int K, int act, hipStream_t s) {
    int mBlocks = M / 128;                 // all call sites have M % 128 == 0
    int grpsN   = (N + 63) / 64;
    wmma_gemm_bf16<<<mBlocks * grpsN, 256, 0, s>>>(A, lda, W, ldw, C, ldc,
                                                   bias, M, N, K, act);
}

static inline void launch_cvt(const float* src, long ss, unsigned short* dst,
                              long ds, long rows, long cols, hipStream_t s) {
    long total = rows * cols;
    int blocks = (int)((total + 255) / 256);
    cvt2d_bf16<<<blocks, 256, 0, s>>>(src, ss, dst, ds, rows, cols);
}

extern "C" void kernel_launch(void* const* d_in, const int* in_sizes, int n_in,
                              void* d_out, int out_size, void* d_ws, size_t ws_size,
                              hipStream_t stream) {
    const float* enc       = (const float*)d_in[0];
    const int*   captions  = (const int*)  d_in[1];
    const float* emb       = (const float*)d_in[2];
    const float* mapper_w  = (const float*)d_in[3];
    const float* mapper_b  = (const float*)d_in[4];
    const float* align_w   = (const float*)d_in[5];
    const float* align_b   = (const float*)d_in[6];
    const float* pre_attn  = (const float*)d_in[7];
    const float* w_ih      = (const float*)d_in[8];
    const float* w_hh      = (const float*)d_in[9];
    const float* b_ih      = (const float*)d_in[10];
    const float* b_hh      = (const float*)d_in[11];
    const float* out_w     = (const float*)d_in[12];
    const float* out_b     = (const float*)d_in[13];
    float* out = (float*)d_out;

    // ---- workspace carve-up (deterministic pointer math) ----
    uint8_t* w = (uint8_t*)d_ws;
    auto take = [&](size_t bytes) -> void* {
        void* p = (void*)w;
        w += (bytes + 255) & ~(size_t)255;
        return p;
    };
    unsigned short* enc_bf    = (unsigned short*)take((size_t)BB * T_ENC * ENC_H * 2);
    unsigned short* mapw_bf   = (unsigned short*)take((size_t)1024 * 1024 * 2);
    unsigned short* ws_bf     = (unsigned short*)take((size_t)ALIGN_A * 1024 * 2);
    unsigned short* we_bf     = (unsigned short*)take((size_t)ALIGN_A * 1024 * 2);
    unsigned short* wcat_bf   = (unsigned short*)take((size_t)2048 * 2048 * 2);
    unsigned short* outw_bf   = (unsigned short*)take((size_t)VOCAB * HID * 2);
    float*          b_comb    = (float*)take((size_t)2048 * 4);
    float*          mapped    = (float*)take((size_t)BB * 1024 * 4);
    float*          enc_proj  = (float*)take((size_t)BB * T_ENC * ALIGN_A * 4);
    float*          h_st      = (float*)take((size_t)BB * HID * 4);
    float*          c_st      = (float*)take((size_t)BB * HID * 4);
    unsigned short* hc_bf     = (unsigned short*)take((size_t)BB * 1024 * 2);
    unsigned short* a_cat     = (unsigned short*)take((size_t)BB * 2048 * 2);
    float*          salign    = (float*)take((size_t)BB * ALIGN_A * 4);
    float*          gates     = (float*)take((size_t)BB * 2048 * 4);
    float*          logits    = (float*)take((size_t)BB * VOCAB * 4);

    // ---- one-time conversions ----
    launch_cvt(enc,      ENC_H, enc_bf,  ENC_H, (long)BB * T_ENC, ENC_H, stream);
    launch_cvt(mapper_w, 1024,  mapw_bf, 1024,  1024, 1024, stream);
    launch_cvt(align_w,         2048, ws_bf, 1024, ALIGN_A, 1024, stream);   // w_s
    launch_cvt(align_w + 1024,  2048, we_bf, 1024, ALIGN_A, 1024, stream);   // w_e
    launch_cvt(w_ih, 1536, wcat_bf,        2048, 2048, 1536, stream);        // Wcat[:, :1536]
    launch_cvt(w_hh, 512,  wcat_bf + 1536, 2048, 2048, 512,  stream);        // Wcat[:, 1536:]
    launch_cvt(out_w, HID, outw_bf, HID, VOCAB, HID, stream);
    add_vec<<<(2048 + 255) / 256, 256, 0, stream>>>(b_ih, b_hh, b_comb, 2048);

    // ---- mapped = relu(enc[:, -1] @ mapper_w^T + mapper_b) ----
    launch_gemm(enc_bf + (long)(T_ENC - 1) * ENC_H, (long)T_ENC * ENC_H,
                mapw_bf, 1024, mapped, 1024, mapper_b,
                BB, 1024, 1024, /*relu*/1, stream);

    // ---- enc_proj = enc @ w_e^T + align_b ----
    launch_gemm(enc_bf, ENC_H, we_bf, 1024, enc_proj, ALIGN_A, align_b,
                BB * T_ENC, ALIGN_A, 1024, 0, stream);

    init_state<<<(BB * HID + 255) / 256, 256, 0, stream>>>(mapped, h_st, c_st,
                                                           hc_bf, a_cat);

    // ---- recurrent decode ----
    for (int t = 0; t < NSTEP; ++t) {
        // salign = hc @ w_s^T
        launch_gemm(hc_bf, 1024, ws_bf, 1024, salign, ALIGN_A, nullptr,
                    BB, ALIGN_A, 1024, 0, stream);

        attention_step<<<BB, 256, 0, stream>>>(enc, enc_proj, salign, pre_attn,
                                               emb, captions, t, a_cat);

        // gates = [ctx,emb,h] @ Wcat^T + (b_ih + b_hh)
        launch_gemm(a_cat, 2048, wcat_bf, 2048, gates, 2048, b_comb,
                    BB, 2048, 2048, 0, stream);

        lstm_cell<<<(BB * HID + 255) / 256, 256, 0, stream>>>(gates, h_st, c_st,
                                                              hc_bf, a_cat);

        // logits = h_new @ out_w^T + out_b   (A = h slot of a_cat, lda=2048)
        launch_gemm(a_cat + 1536, 2048, outw_bf, HID, logits, VOCAB, out_b,
                    BB, VOCAB, HID, 0, stream);

        log_softmax_out<<<BB, 256, 0, stream>>>(logits, out, t);
    }
}
